// Transformer_66760971649264
// MI455X (gfx1250) — compile-verified, matrix operand order checked by reading, not run on hardware
//
#include <hip/hip_runtime.h>
#include <math.h>

// ---------------- constants ----------------
#define QN   300
#define BN   2
#define SN   4096
#define DN   256
#define HN   8
#define HDN  32
#define DFFN 2048
#define TN   (QN*BN)     // 600 tokens
#define TSN  (SN*BN)     // 8192 memory tokens
#define BHN  (BN*HN)     // 16 head-batches
#define SQPAD 320        // Q=300 padded to multiple of 32 (for transposed V)

typedef _Float16 f16;
typedef _Float16 v16h __attribute__((ext_vector_type(16)));
typedef float    v8f  __attribute__((ext_vector_type(8)));
typedef float    v4f  __attribute__((ext_vector_type(4)));

__device__ __forceinline__ v4f ld4(const float* p) { return *(const v4f*)p; }

// Assemble a 16-element f16 fragment from two contiguous 8-float runs.
// Element order matches ISA 7.12.2 16-bit A/B layout: for lane-half kh,
// elements 0..7 hold k = kh*8+0..7, elements 8..15 hold k = 16+kh*8+0..7.
__device__ __forceinline__ v16h pack16(v4f a0, v4f a1, v4f a2, v4f a3) {
  v16h r;
  r[0] = (f16)a0[0];  r[1] = (f16)a0[1];  r[2]  = (f16)a0[2];  r[3]  = (f16)a0[3];
  r[4] = (f16)a1[0];  r[5] = (f16)a1[1];  r[6]  = (f16)a1[2];  r[7]  = (f16)a1[3];
  r[8] = (f16)a2[0];  r[9] = (f16)a2[1];  r[10] = (f16)a2[2];  r[11] = (f16)a2[3];
  r[12]= (f16)a3[0];  r[13]= (f16)a3[1];  r[14] = (f16)a3[2];  r[15] = (f16)a3[3];
  return r;
}

// ---------------- generic WMMA GEMM ----------------
// C[M,N] = act( (A(+A2))[M,K] @ W[N,K]^T + bias[N] ) * oscale
// mode 0: token-major C[m*N+n]
// mode 1: per-head    C[(bh*Smax + q)*32 + hd]         (Q/K layout)
// mode 2: head-transposed C[(bh*32 + hd)*Smax + q]     (V layout, Smax padded)
__global__ __launch_bounds__(32)
void gemm16(const float* __restrict__ A, const float* __restrict__ A2,
            const float* __restrict__ W, const float* __restrict__ bias,
            float* __restrict__ C,
            int M, int N, int K, int mode, int Bdim, int Smax,
            float oscale, int relu)
{
  int lane = threadIdx.x & 31;
  int lm = lane & 15, kh = lane >> 4;
  int mbase = blockIdx.x * 16;
  int nbase = blockIdx.y * 16;
  // Clamped rows: OOB A-rows / W-rows read valid memory; results masked on store.
  int amc = mbase + lm; if (amc > M - 1) amc = M - 1;
  int wnc = nbase + lm; if (wnc > N - 1) wnc = N - 1;

  const bool hasA2 = (A2 != nullptr);          // SGPR test -> scalar branch
  const float* arow  = A + (size_t)amc * K + kh * 8;
  const float* a2row = (hasA2 ? A2 : A) + (size_t)amc * K + kh * 8;
  const float* wrow  = W + (size_t)wnc * K + kh * 8;

  v8f acc = {0.f,0.f,0.f,0.f,0.f,0.f,0.f,0.f};
#pragma unroll 2
  for (int kt = 0; kt < K; kt += 32) {
    v4f a0 = ld4(arow + kt),      a1 = ld4(arow + kt + 4);
    v4f a2 = ld4(arow + kt + 16), a3 = ld4(arow + kt + 20);
    if (hasA2) {                         // wave-uniform branch
      a0 += ld4(a2row + kt);      a1 += ld4(a2row + kt + 4);
      a2 += ld4(a2row + kt + 16); a3 += ld4(a2row + kt + 20);
    }
    v4f w0 = ld4(wrow + kt),      w1 = ld4(wrow + kt + 4);
    v4f w2 = ld4(wrow + kt + 16), w3 = ld4(wrow + kt + 20);
    v16h af = pack16(a0, a1, a2, a3);
    v16h bf = pack16(w0, w1, w2, w3);
    acc = __builtin_amdgcn_wmma_f32_16x16x32_f16(false, af, false, bf,
                                                 (short)0, acc, false, false);
  }

#pragma unroll
  for (int r = 0; r < 8; ++r) {
    int m = mbase + r + kh * 8;
    int n = nbase + lm;
    if (m < M && n < N) {
      float v = (acc[r] + bias[n]) * oscale;
      if (relu) v = v > 0.f ? v : 0.f;
      if (mode == 0) {
        C[(size_t)m * N + n] = v;
      } else {
        int q = m / Bdim, b = m % Bdim;
        int h = n >> 5, hd = n & 31;
        int bh = b * HN + h;
        if (mode == 1) C[((size_t)bh * Smax + q) * HDN + hd] = v;
        else           C[((size_t)bh * HDN + hd) * (size_t)Smax + q] = v;
      }
    }
  }
}

// ---------------- flash attention (online softmax), optional Gaussian bias ----
// qp/kp: [BH][Sx][32]; vp: [BH][32][SkvPad] (transposed, padded/zero-filled)
// out token-major [Sq*B, 256]
__global__ __launch_bounds__(32)
void flash_attn(const float* __restrict__ qp, const float* __restrict__ kp,
                const float* __restrict__ vp, float* __restrict__ outTok,
                int Sq, int Skv, int SkvPad, int Bdim, int bias_mode,
                const float* __restrict__ grid,
                const float* __restrict__ point,
                const float* __restrict__ sc2)
{
  __shared__ float p_lds[16 * 32];
  int lane = threadIdx.x & 31;
  int lm = lane & 15, kh = lane >> 4;
  int bh = blockIdx.y;
  int qbase = blockIdx.x * 16;

  // Q fragment (HD==32 -> exactly one K slab); clamp OOB q rows.
  v16h qf;
  {
    int qc = qbase + lm; if (qc > Sq - 1) qc = Sq - 1;
    const float* qrow = qp + ((size_t)bh * Sq + qc) * HDN + kh * 8;
    qf = pack16(ld4(qrow), ld4(qrow + 4), ld4(qrow + 16), ld4(qrow + 20));
  }

  float px[8], py[8], sx[8], sy[8];
#pragma unroll
  for (int r = 0; r < 8; ++r) { px[r]=py[r]=sx[r]=sy[r]=0.f; }
  if (bias_mode) {
#pragma unroll
    for (int r = 0; r < 8; ++r) {
      int qc = qbase + r + kh * 8; if (qc > Sq - 1) qc = Sq - 1;
      size_t pidx = ((size_t)qc * 16 + bh) * 2;
      px[r] = point[pidx]; py[r] = point[pidx + 1];
      sx[r] = sc2[pidx];   sy[r] = sc2[pidx + 1];
    }
  }

  float mi[8], li[8];
#pragma unroll
  for (int r = 0; r < 8; ++r) { mi[r] = -3.0e38f; li[r] = 0.f; }
  v8f o0 = {0.f,0.f,0.f,0.f,0.f,0.f,0.f,0.f};
  v8f o1 = {0.f,0.f,0.f,0.f,0.f,0.f,0.f,0.f};

  for (int st = 0; st < Skv; st += 32) {
    // ---- logits: two 16-col WMMAs; clamp OOB s rows ----
    v8f lacc[2];
#pragma unroll
    for (int j = 0; j < 2; ++j) {
      int sc = st + j * 16 + lm; if (sc > Skv - 1) sc = Skv - 1;
      const float* krow = kp + ((size_t)bh * Skv + sc) * HDN + kh * 8;
      v16h kf = pack16(ld4(krow), ld4(krow + 4), ld4(krow + 16), ld4(krow + 20));
      v8f z = {0.f,0.f,0.f,0.f,0.f,0.f,0.f,0.f};
      lacc[j] = __builtin_amdgcn_wmma_f32_16x16x32_f16(false, qf, false, kf,
                                                       (short)0, z, false, false);
    }

    // ---- fused Gaussian bias + OOB mask (selects, no branches) ----
    float lg[2][8];
#pragma unroll
    for (int j = 0; j < 2; ++j) {
      int n = st + j * 16 + lm;
      int nc = n > Skv - 1 ? Skv - 1 : n;
      float gx = 0.f, gy = 0.f;
      if (bias_mode) { gx = grid[nc * 2]; gy = grid[nc * 2 + 1]; }
#pragma unroll
      for (int r = 0; r < 8; ++r) {
        float v = lacc[j][r];
        if (bias_mode) {
          float dx = px[r] - gx, dy = py[r] - gy;
          float d = dx * dx * sx[r] + dy * dy * sy[r];
          v -= fabsf(d) * 0.125f;   // -|dist|/SMOOTH
        }
        lg[j][r] = (n < Skv) ? v : -3.0e38f;
      }
    }

    // ---- online softmax update (row reductions across a half-wave) ----
#pragma unroll
    for (int r = 0; r < 8; ++r) {
      float rm = fmaxf(lg[0][r], lg[1][r]);
      rm = fmaxf(rm, __shfl_xor(rm, 1, 32));
      rm = fmaxf(rm, __shfl_xor(rm, 2, 32));
      rm = fmaxf(rm, __shfl_xor(rm, 4, 32));
      rm = fmaxf(rm, __shfl_xor(rm, 8, 32));
      float nm = fmaxf(mi[r], rm);
      float scale = __expf(mi[r] - nm);
      float p0 = __expf(lg[0][r] - nm);
      float p1 = __expf(lg[1][r] - nm);
      float rs = p0 + p1;
      rs += __shfl_xor(rs, 1, 32);
      rs += __shfl_xor(rs, 2, 32);
      rs += __shfl_xor(rs, 4, 32);
      rs += __shfl_xor(rs, 8, 32);
      li[r] = li[r] * scale + rs;
      mi[r] = nm;
      o0[r] *= scale;
      o1[r] *= scale;
      // stage P tile in LDS for layout change (D-frag -> A-frag)
      p_lds[(r + kh * 8) * 32 + lm]      = p0;
      p_lds[(r + kh * 8) * 32 + 16 + lm] = p1;
    }
    __syncthreads();

    // ---- O += P @ V  (V transposed: contiguous along S -> b128 loads) ----
    v16h pf;
#pragma unroll
    for (int e = 0; e < 16; ++e) {
      int v = e >> 1;
      int ks = ((v & 4) << 2) + kh * 8 + ((v & 3) << 1) + (e & 1);
      pf[e] = (f16)p_lds[lm * 32 + ks];
    }
#pragma unroll
    for (int j = 0; j < 2; ++j) {
      const float* vrow = vp + ((size_t)(bh * HDN + j * 16 + lm)) * SkvPad
                             + st + kh * 8;
      v16h vf = pack16(ld4(vrow), ld4(vrow + 4), ld4(vrow + 16), ld4(vrow + 20));
      if (j == 0)
        o0 = __builtin_amdgcn_wmma_f32_16x16x32_f16(false, pf, false, vf,
                                                    (short)0, o0, false, false);
      else
        o1 = __builtin_amdgcn_wmma_f32_16x16x32_f16(false, pf, false, vf,
                                                    (short)0, o1, false, false);
    }
    __syncthreads();
  }

  // ---- normalize + write token-major [q*B+b][h*32+hd] ----
  int b = bh >> 3, h = bh & 7;
#pragma unroll
  for (int r = 0; r < 8; ++r) {
    int q = qbase + r + kh * 8;
    if (q < Sq) {
      float inv = li[r] > 0.f ? 1.f / li[r] : 0.f;
      size_t base = ((size_t)(q * Bdim + b)) * DN + h * HDN;
      outTok[base + lm]      = o0[r] * inv;
      outTok[base + 16 + lm] = o1[r] * inv;
    }
  }
}

// ---------------- LayerNorm over D=256 of (A + B) ----------------
__global__ void add_ln(const float* __restrict__ A, const float* __restrict__ Badd,
                       const float* __restrict__ g, const float* __restrict__ be,
                       float* __restrict__ out)
{
  __shared__ float red[256];
  int t = blockIdx.x;
  int i = threadIdx.x;
  float x = A[(size_t)t * DN + i] + (Badd ? Badd[(size_t)t * DN + i] : 0.f);
  red[i] = x;
  __syncthreads();
  for (int off = 128; off > 0; off >>= 1) {
    if (i < off) red[i] += red[i + off];
    __syncthreads();
  }
  float mean = red[0] * (1.f / DN);
  __syncthreads();
  float d = x - mean;
  red[i] = d * d;
  __syncthreads();
  for (int off = 128; off > 0; off >>= 1) {
    if (i < off) red[i] += red[i + off];
    __syncthreads();
  }
  float var = red[0] * (1.f / DN);
  out[(size_t)t * DN + i] = d * rsqrtf(var + 1e-5f) * g[i] + be[i];
}

// ---------------- x1 + masked query_pos ----------------
__global__ void mask_add(const float* __restrict__ x1, const float* __restrict__ qpos,
                         const int* __restrict__ ro1, const int* __restrict__ ro2,
                         float* __restrict__ out)
{
  int t = blockIdx.x;
  int i = threadIdx.x;
  int q = t >> 1, b = t & 1;
  const int* ro = b ? ro2 : ro1;
  bool msk = (ro[q * 4] == 0) | (ro[q * 4 + 1] == 0) |
             (ro[q * 4 + 2] == 0) | (ro[q * 4 + 3] == 0);
  float add = msk ? 0.f : qpos[(size_t)t * DN + i];
  out[(size_t)t * DN + i] = x1[(size_t)t * DN + i] + add;
}

// ---------------- ref / point / scale^2 ----------------
__global__ void refpoint(const float* __restrict__ ri, const float* __restrict__ hw,
                         const float* __restrict__ po, const float* __restrict__ sraw,
                         float* __restrict__ ref_out, float* __restrict__ point,
                         float* __restrict__ sc2, int n)
{
  int idx = blockIdx.x * 256 + threadIdx.x;
  if (idx >= n) return;
  int t = idx >> 4, k = idx & 15;
  int b = t & 1, c = k & 1;
  float r = ri[t * 2 + c];
  float refv = hw[b * 2 + c] * (1.f / (1.f + __expf(-r))) * (1.f / 32.f);
  ref_out[idx] = refv;
  point[idx] = refv + po[idx];
  float s = sraw[idx];
  sc2[idx] = s * s;
}

// ---------------- host ----------------
extern "C" void kernel_launch(void* const* d_in, const int* in_sizes, int n_in,
                              void* d_out, int out_size, void* d_ws, size_t ws_size,
                              hipStream_t stream) {
  (void)in_sizes; (void)n_in; (void)out_size; (void)ws_size;
  const float* grid   = (const float*)d_in[0];
  const float* h_w    = (const float*)d_in[1];
  const float* tgt    = (const float*)d_in[2];
  const float* memory = (const float*)d_in[3];
  const float* pos    = (const float*)d_in[4];
  const float* qpos   = (const float*)d_in[5];
  const int*   ro1    = (const int*)d_in[6];
  const int*   ro2    = (const int*)d_in[7];
  const float* sa_Wi = (const float*)d_in[8];
  const float* sa_bi = (const float*)d_in[9];
  const float* sa_Wo = (const float*)d_in[10];
  const float* sa_bo = (const float*)d_in[11];
  const float* ca_Wi = (const float*)d_in[12];
  const float* ca_bi = (const float*)d_in[13];
  const float* ca_Wo = (const float*)d_in[14];
  const float* ca_bo = (const float*)d_in[15];
  const float* W1 = (const float*)d_in[16];
  const float* b1 = (const float*)d_in[17];
  const float* W2 = (const float*)d_in[18];
  const float* b2 = (const float*)d_in[19];
  const float* n1g = (const float*)d_in[20]; const float* n1b = (const float*)d_in[21];
  const float* n2g = (const float*)d_in[22]; const float* n2b = (const float*)d_in[23];
  const float* n3g = (const float*)d_in[24]; const float* n3b = (const float*)d_in[25];
  const float* n4g = (const float*)d_in[26]; const float* n4b = (const float*)d_in[27];
  const float* p1_W0 = (const float*)d_in[28]; const float* p1_b0 = (const float*)d_in[29];
  const float* p1_W1 = (const float*)d_in[30]; const float* p1_b1 = (const float*)d_in[31];
  const float* p1_W2 = (const float*)d_in[32]; const float* p1_b2 = (const float*)d_in[33];
  const float* p2W = (const float*)d_in[34]; const float* p2b = (const float*)d_in[35];
  const float* p3W = (const float*)d_in[36]; const float* p3b = (const float*)d_in[37];

  // workspace carve-up (floats)
  float* w = (float*)d_ws;
  size_t o = 0;
  auto alloc = [&](size_t n) { float* p = w + o; o += n; return p; };
  float* out_n4   = alloc((size_t)TN * DN);
  float* point_off= alloc((size_t)TN * 16);
  float* sraw     = alloc((size_t)TN * 16);
  float* sc2      = alloc((size_t)TN * 16);
  float* pointb   = alloc((size_t)TN * 16);
  float* proj_q   = alloc((size_t)BHN * QN * HDN);
  float* proj_k   = alloc((size_t)BHN * QN * HDN);
  float* proj_v   = alloc((size_t)BHN * HDN * SQPAD);   // transposed V, padded
  float* ao       = alloc((size_t)TN * DN);
  float* tmpD     = alloc((size_t)TN * DN);
  float* x1       = alloc((size_t)TN * DN);
  float* x2       = alloc((size_t)TN * DN);
  float* h1b      = alloc((size_t)TN * DN);
  float* h2b      = alloc((size_t)TN * DN);
  float* qk3      = alloc((size_t)TN * DN);
  float* kp_ca    = alloc((size_t)BHN * SN * HDN);
  float* vp_ca    = alloc((size_t)BHN * HDN * SN);      // transposed V (S mult of 32)
  float* ao3      = alloc((size_t)TN * DN);
  float* ffh      = alloc((size_t)TN * DFFN);

  float* out_x    = (float*)d_out;               // [Q,B,D]
  float* out_ri   = out_x + (size_t)TN * DN;     // [Q,B,2]
  float* out_ref  = out_ri + (size_t)TN * 2;     // [Q,B,16]
  float* out_tgt3 = out_ref + (size_t)TN * 16;   // [Q,B,D]

  const float qscale = 0.17677669529663687f;     // HD^-0.5
  dim3 blk(32);
  auto gg = [](int M, int N) { return dim3((M + 15) / 16, (N + 15) / 16); };
  const int QT = (QN + 15) / 16;                 // 19 q tiles

  // 1. out = ln(tgt + query_pos) [n4]
  add_ln<<<TN, 256, 0, stream>>>(tgt, qpos, n4g, n4b, out_n4);
  // 2-3. point_offset, raw scale
  gemm16<<<gg(TN, 16), blk, 0, stream>>>(out_n4, nullptr, p2W, p2b, point_off,
                                         TN, 16, DN, 0, BN, 0, 1.f, 0);
  gemm16<<<gg(TN, 16), blk, 0, stream>>>(out_n4, nullptr, p3W, p3b, sraw,
                                         TN, 16, DN, 0, BN, 0, 1.f, 0);
  // 4. SA projections (per-head layouts); zero-fill padded V first
  hipMemsetAsync(proj_v, 0, (size_t)BHN * HDN * SQPAD * sizeof(float), stream);
  gemm16<<<gg(TN, DN), blk, 0, stream>>>(tgt, qpos, sa_Wi, sa_bi, proj_q,
                                         TN, DN, DN, 1, BN, QN, qscale, 0);
  gemm16<<<gg(TN, DN), blk, 0, stream>>>(tgt, qpos, sa_Wi + DN * DN, sa_bi + DN, proj_k,
                                         TN, DN, DN, 1, BN, QN, 1.f, 0);
  gemm16<<<gg(TN, DN), blk, 0, stream>>>(tgt, nullptr, sa_Wi + 2 * DN * DN, sa_bi + 2 * DN, proj_v,
                                         TN, DN, DN, 2, BN, SQPAD, 1.f, 0);
  // 5. SA flash attention
  flash_attn<<<dim3(QT, BHN), blk, 0, stream>>>(proj_q, proj_k, proj_v, ao,
                                                QN, QN, SQPAD, BN, 0,
                                                nullptr, nullptr, nullptr);
  // 6-7. SA out proj + ln
  gemm16<<<gg(TN, DN), blk, 0, stream>>>(ao, nullptr, sa_Wo, sa_bo, tmpD,
                                         TN, DN, DN, 0, BN, 0, 1.f, 0);
  add_ln<<<TN, 256, 0, stream>>>(tgt, tmpD, n1g, n1b, x1);
  // 8. point1 MLP -> ref_inter (output)
  gemm16<<<gg(TN, DN), blk, 0, stream>>>(out_n4, nullptr, p1_W0, p1_b0, h1b,
                                         TN, DN, DN, 0, BN, 0, 1.f, 1);
  gemm16<<<gg(TN, DN), blk, 0, stream>>>(h1b, nullptr, p1_W1, p1_b1, h2b,
                                         TN, DN, DN, 0, BN, 0, 1.f, 1);
  gemm16<<<gg(TN, 2), blk, 0, stream>>>(h2b, nullptr, p1_W2, p1_b2, out_ri,
                                        TN, 2, DN, 0, BN, 0, 1.f, 0);
  // 9. ref (output) / point / scale^2
  refpoint<<<(TN * 16 + 255) / 256, 256, 0, stream>>>(out_ri, h_w, point_off, sraw,
                                                      out_ref, pointb, sc2, TN * 16);
  // 10. qk3 = x1 + masked query_pos
  mask_add<<<TN, 256, 0, stream>>>(x1, qpos, ro1, ro2, qk3);
  // 11. CA projections: K/V over 8192 memory tokens, two Q variants
  gemm16<<<gg(TSN, DN), blk, 0, stream>>>(memory, pos, ca_Wi + DN * DN, ca_bi + DN, kp_ca,
                                          TSN, DN, DN, 1, BN, SN, 1.f, 0);
  gemm16<<<gg(TSN, DN), blk, 0, stream>>>(memory, nullptr, ca_Wi + 2 * DN * DN, ca_bi + 2 * DN, vp_ca,
                                          TSN, DN, DN, 2, BN, SN, 1.f, 0);
  gemm16<<<gg(TN, DN), blk, 0, stream>>>(x1, qpos, ca_Wi, ca_bi, proj_q,
                                         TN, DN, DN, 1, BN, QN, qscale, 0);
  gemm16<<<gg(TN, DN), blk, 0, stream>>>(qk3, nullptr, ca_Wi, ca_bi, proj_k,
                                         TN, DN, DN, 1, BN, QN, qscale, 0);
  // 12. CA flash attention with fused Gaussian bias (x2)
  flash_attn<<<dim3(QT, BHN), blk, 0, stream>>>(proj_q, kp_ca, vp_ca, ao,
                                                QN, SN, SN, BN, 1, grid, pointb, sc2);
  flash_attn<<<dim3(QT, BHN), blk, 0, stream>>>(proj_k, kp_ca, vp_ca, ao3,
                                                QN, SN, SN, BN, 1, grid, pointb, sc2);
  // 13. CA out projections (tgt3 goes straight to output)
  gemm16<<<gg(TN, DN), blk, 0, stream>>>(ao, nullptr, ca_Wo, ca_bo, tmpD,
                                         TN, DN, DN, 0, BN, 0, 1.f, 0);
  gemm16<<<gg(TN, DN), blk, 0, stream>>>(ao3, nullptr, ca_Wo, ca_bo, out_tgt3,
                                         TN, DN, DN, 0, BN, 0, 1.f, 0);
  // 14. x2 = ln(x1 + tgt2)
  add_ln<<<TN, 256, 0, stream>>>(x1, tmpD, n2g, n2b, x2);
  // 15-17. FFN + final ln -> output x
  gemm16<<<gg(TN, DFFN), blk, 0, stream>>>(x2, nullptr, W1, b1, ffh,
                                           TN, DFFN, DN, 0, BN, 0, 1.f, 1);
  gemm16<<<gg(TN, DN), blk, 0, stream>>>(ffh, nullptr, W2, b2, tmpD,
                                         TN, DN, DFFN, 0, BN, 0, 1.f, 0);
  add_ln<<<TN, 256, 0, stream>>>(x2, tmpD, n3g, n3b, out_x);
}
